// MDMA_79955111182457
// MI455X (gfx1250) — matrix operand order repaired and checked by reading, not compiled, not run on hardware
//
#include <hip/hip_runtime.h>
#include <hip/hip_bf16.h>

// ---------------------------------------------------------------------------
// MDMA density model for gfx1250 (MI455X):
//   stage 1: per-(b,d,m) R=3 MLP  -> phidots [B,32,1000]   (fp32 VALU)
//   stage 2: stabilizer fm[b] + first product-pool -> T0 [B,16,1000]
//   stage 3: column sums of softplus(aht_l) for mixture normalization
//   stage 4: per level: prep_b (normalized bf16 B panels, pre-swizzled into
//            WMMA fragment layout), prep_a (pooled bf16 A, K padded to 1024),
//            then batched GEMM via v_wmma_f32_16x16x32_bf16 with a
//            double-buffered global_load_async_to_lds pipeline
//   stage 5: final weighted reduction -> log lk - 32*fm
// ---------------------------------------------------------------------------

#define MDIM 1000
#define KPAD 1024
#define DDIM 32
#define BDIM 512

typedef __attribute__((ext_vector_type(16))) __bf16          v16bf;
typedef __attribute__((ext_vector_type(8)))  float           v8f;

#if defined(__gfx1250__) && __has_builtin(__builtin_amdgcn_global_load_async_to_lds_b128)
#define USE_ASYNC_LDS 1
// exact pointee type expected by the builtin: int __vector_size__(16)
typedef int async_i4 __attribute__((vector_size(16)));
typedef __attribute__((address_space(1))) async_i4 gas_i4;
typedef __attribute__((address_space(3))) async_i4 las_i4;
#else
#define USE_ASYNC_LDS 0
#endif

__device__ __forceinline__ void wait_async0() {
#if defined(__gfx1250__)
#if __has_builtin(__builtin_amdgcn_s_wait_asynccnt)
    __builtin_amdgcn_s_wait_asynccnt(0);
#else
    asm volatile("s_wait_asynccnt 0x0" ::: "memory");
#endif
#endif
}

__device__ __forceinline__ unsigned short f2bf(float f) {
    // round-to-nearest-even fp32 -> bf16 (inputs here are finite)
    unsigned int u = __builtin_bit_cast(unsigned int, f);
    unsigned int r = u + 0x7FFFu + ((u >> 16) & 1u);
    return (unsigned short)(r >> 16);
}

__device__ __forceinline__ float nonneg(float x) {
    // softplus(10x)/10 = max(x,0) + log1p(exp(-10|x|))/10
    return fmaxf(x, 0.f) + 0.1f * __logf(1.f + __expf(-10.f * fabsf(x)));
}

__device__ __forceinline__ float fast_tanh(float x) {
    float t = __expf(-2.f * fabsf(x));
    float r = (1.f - t) / (1.f + t);
    return copysignf(r, x);
}

// ---------------------------------------------------------------- stage 1 ---
__global__ __launch_bounds__(256)
void mlp_phidots(const float* __restrict__ X,
                 const float* __restrict__ w0, const float* __restrict__ b0,
                 const float* __restrict__ a0,
                 const float* __restrict__ w1, const float* __restrict__ b1,
                 const float* __restrict__ a1,
                 const float* __restrict__ w2, const float* __restrict__ b2,
                 float* __restrict__ P) {
    int dm = blockIdx.x * 256 + threadIdx.x;       // 0 .. 31999
    int b  = blockIdx.y;
    if (dm >= DDIM * MDIM) return;
    int d = dm / MDIM;
    float x = X[b * DDIM + d];
    size_t o3 = (size_t)dm * 3;

    float phi[3], pd[3];
#pragma unroll
    for (int r = 0; r < 3; ++r) {
        float wn  = nonneg(w0[o3 + r]);
        float pre = x * wn + b0[o3 + r];
        float th  = fast_tanh(pre);
        float ta  = fast_tanh(a0[o3 + r]);
        pd[r]  = wn * (1.f + ta * (1.f - th * th));
        phi[r] = pre + th * ta;
    }
    float phi1[3], pd1[3];
#pragma unroll
    for (int l = 0; l < 3; ++l) {
        float pre = b1[o3 + l], acc = 0.f;
#pragma unroll
        for (int r = 0; r < 3; ++r) {
            float wn = nonneg(w1[(o3 + r) * 3 + l]);
            pre += phi[r] * wn;
            acc += pd[r] * wn;
        }
        float th = fast_tanh(pre);
        float ta = fast_tanh(a1[o3 + l]);
        pd1[l]  = acc * (1.f + ta * (1.f - th * th));
        phi1[l] = pre + th * ta;
    }
    float pre = b2[dm], acc = 0.f;
#pragma unroll
    for (int r = 0; r < 3; ++r) {
        float wn = nonneg(w2[o3 + r]);
        pre += phi1[r] * wn;
        acc += pd1[r] * wn;
    }
    float s = 1.f / (1.f + __expf(-pre));
    P[((size_t)b * DDIM + d) * MDIM + (dm % MDIM)] = acc * s * (1.f - s);
}

// ---------------------------------------------------------------- stage 2 ---
__global__ __launch_bounds__(256)
void stab_t0(const float* __restrict__ P, float* __restrict__ T0,
             float* __restrict__ fmOut) {
    int b = blockIdx.x, tid = threadIdx.x;
    int lane = tid & 31, wv = tid >> 5;
    __shared__ float shred[16 * 8];
    __shared__ float fmsh;

    const float* Pb = P + (size_t)b * DDIM * MDIM;
    float mn[16];
#pragma unroll
    for (int k = 0; k < 16; ++k) mn[k] = 3.4e38f;

    for (int m = tid; m < MDIM; m += 256) {
#pragma unroll
        for (int k = 0; k < 16; ++k) {
            float v0 = fmaxf(Pb[(2 * k) * MDIM + m], 1e-40f);
            float v1 = fmaxf(Pb[(2 * k + 1) * MDIM + m], 1e-40f);
            float nl = -0.5f * (__logf(v0) + __logf(v1));
            mn[k] = fminf(mn[k], nl);
        }
    }
#pragma unroll
    for (int k = 0; k < 16; ++k) {
        float v = mn[k];
#pragma unroll
        for (int off = 16; off > 0; off >>= 1)
            v = fminf(v, __shfl_xor(v, off));
        if (lane == 0) shred[k * 8 + wv] = v;
    }
    __syncthreads();
    if (tid == 0) {
        float s = 0.f;
#pragma unroll
        for (int k = 0; k < 16; ++k) {
            float v = 3.4e38f;
            for (int i = 0; i < 8; ++i) v = fminf(v, shred[k * 8 + i]);
            s += v;
        }
        float fm = s * (1.f / 16.f);
        fmsh = fm;
        fmOut[b] = fm;
    }
    __syncthreads();
    float e2 = __expf(2.f * fmsh);
    for (int idx = tid; idx < 16 * MDIM; idx += 256) {
        int k = idx / MDIM, m = idx - k * MDIM;
        float v = Pb[(2 * k) * MDIM + m] * Pb[(2 * k + 1) * MDIM + m] * e2;
        T0[((size_t)b * 16 + k) * MDIM + m] = v;
    }
}

// ---------------------------------------------------------------- stage 3 ---
__global__ __launch_bounds__(256)
void colsum_inv(const float* __restrict__ W, float* __restrict__ inv) {
    int n = blockIdx.x * 256 + threadIdx.x;
    int g = blockIdx.y;
    if (n >= MDIM) return;
    const float* wg = W + (size_t)g * MDIM * MDIM;
    float s = 0.f;
    for (int m = 0; m < MDIM; ++m) s += nonneg(wg[(size_t)m * MDIM + n]);
    inv[g * MDIM + n] = 1.f / s;
}

// --------------------------------------------------------------- prep B ----
// One block builds one 32(K)x64(N) panel of nonneg(W)*invcol in bf16,
// stored contiguously (4KB) in the B-fragment register layout ht_gemm uses (contiguous k order):
//   fragment j = n-subtile of 16; "lane" bl = (n%16) + 16*(k/16);
//   slot = k%16 (K ascending). K beyond 1000 zero-padded (KPAD=1024).
__global__ __launch_bounds__(256)
void prep_b(const float* __restrict__ W, const float* __restrict__ inv,
            unsigned short* __restrict__ panels) {
    const int nt = blockIdx.x;          // 0..15
    const int kt = blockIdx.y;          // 0..31  (K padded to 1024)
    const int g  = blockIdx.z;
    const int tid = threadIdx.x;
    const int k0 = kt * 32, n0 = nt * 64;
    unsigned short* pan = panels + (((size_t)g * 16 + nt) * 32 + kt) * 2048;
    const float* wg = W + (size_t)g * MDIM * MDIM;
    const float* ig = inv + g * MDIM;
#pragma unroll
    for (int i = 0; i < 8; ++i) {
        int e  = i * 256 + tid;         // 0..2047
        int kk = e >> 6, nn = e & 63;
        int kg = k0 + kk, ng = n0 + nn;
        float val = 0.f;
        if (kg < MDIM && ng < MDIM)
            val = nonneg(wg[(size_t)kg * MDIM + ng]) * ig[ng];
        int j = nn >> 4, bl = (nn & 15) + ((kk >> 4) << 4), slot = kk & 15;
        pan[(j * 32 + bl) * 16 + slot] = f2bf(val);
    }
}

// --------------------------------------------------------------- prep A ----
// Pooled (optional) fp32 -> bf16 A rows, K padded to 1024 with zeros.
// dst layout: [B][G][1024] bf16.
__global__ __launch_bounds__(256)
void prep_a(const float* __restrict__ src, unsigned short* __restrict__ dst,
            int G, int pool) {
    size_t idx = (size_t)blockIdx.x * 256 + threadIdx.x;  // over B*G*1024
    int m = (int)(idx & (KPAD - 1));
    size_t bg = idx >> 10;                                 // b*G + g
    float v = 0.f;
    if (m < MDIM) {
        if (pool) {
            size_t b = bg / (size_t)G;
            int g = (int)(bg - b * G);
            const float* r = src + (b * 2 * G + 2 * g) * (size_t)MDIM;
            v = r[m] * r[MDIM + m];
        } else {
            v = src[bg * (size_t)MDIM + m];
        }
    }
    dst[idx] = f2bf(v);
}

// ---------------------------------------------------------------- stage 4 ---
// One block = 8 waves computes a 128(batch) x 64(n) tile of
//   C[b,n] = sum_m A[b,m] * Bn[g,m,n]   (both operands prepacked bf16)
// Double-buffered pipeline: async DMA of panel kt+1 overlaps the 4 WMMAs
// consuming panel kt; one barrier per k-step. No guards (K padded to 1024).
__global__ __launch_bounds__(256)
void ht_gemm(const unsigned short* __restrict__ Abf,
             const unsigned short* __restrict__ Bp,
             float* __restrict__ C, int G) {
    const int nt    = blockIdx.x;
    const int bBase = blockIdx.y * 128;
    const int g     = blockIdx.z;
    const int tid   = threadIdx.x;
    const int lane  = tid & 31;
    const int wv    = tid >> 5;
    const int laneHi = lane >> 4;

    __shared__ __align__(16) unsigned short ldsB[2][2048];  // 2 x 4KB panels

    v8f acc[4];
#pragma unroll
    for (int j = 0; j < 4; ++j)
#pragma unroll
        for (int r = 0; r < 8; ++r) acc[j][r] = 0.f;

    const int brow = bBase + wv * 16 + (lane & 15);       // A row (batch idx)
    const unsigned short* arow = Abf + ((size_t)brow * G + g) * KPAD;
    const unsigned short* gsrc =
        Bp + ((size_t)g * 16 + nt) * 32 * 2048 + tid * 8; // this thread's 16B

#if USE_ASYNC_LDS
    // prologue: DMA panel 0 into buffer 0
    __builtin_amdgcn_global_load_async_to_lds_b128(
        (gas_i4*)gsrc, (las_i4*)(&ldsB[0][0] + tid * 8), 0, 0);
    wait_async0();
    __syncthreads();
#else
    *(uint4*)(&ldsB[0][0] + tid * 8) = *(const uint4*)gsrc;
    __syncthreads();
#endif

    for (int kt = 0; kt < 32; ++kt) {
        const int cur = kt & 1;
#if USE_ASYNC_LDS
        // issue DMA for next panel into the other buffer (free since the
        // barrier at the end of iteration kt-1)
        if (kt + 1 < 32)
            __builtin_amdgcn_global_load_async_to_lds_b128(
                (gas_i4*)(gsrc + (size_t)(kt + 1) * 2048),
                (las_i4*)(&ldsB[cur ^ 1][0] + tid * 8), 0, 0);
#else
        if (kt + 1 < 32)
            *(uint4*)(&ldsB[cur ^ 1][0] + tid * 8) =
                *(const uint4*)(gsrc + (size_t)(kt + 1) * 2048);
#endif

        // A fragment: K chunk [c0..c0+15] (bf16, direct)
        int c0 = kt * 32 + laneHi * 16;
        union { v16bf bf; uint4 q[2]; } au;
        au.q[0] = *(const uint4*)(arow + c0);
        au.q[1] = *(const uint4*)(arow + c0 + 8);

        // preload all 4 B fragments into distinct registers, then WMMA quad
        uint4 fq[4][2];
#pragma unroll
        for (int j = 0; j < 4; ++j) {
            const uint4* fp = (const uint4*)&ldsB[cur][(j * 32 + lane) * 16];
            fq[j][0] = fp[0];
            fq[j][1] = fp[1];
        }
#pragma unroll
        for (int j = 0; j < 4; ++j) {
            union { v16bf bf; uint4 q[2]; } bu;
            bu.q[0] = fq[j][0];
            bu.q[1] = fq[j][1];
            acc[j] = __builtin_amdgcn_wmma_f32_16x16x32_bf16(
                false, au.bf, false, bu.bf, (short)0, acc[j], false, false);
        }

#if USE_ASYNC_LDS
        wait_async0();                   // next panel (this wave's slice) done
#endif
        __syncthreads();                 // all slices done; all reads retired
    }

    // store: VGPR r, lanes 0-15 -> row r, lanes 16-31 -> row r+8
#pragma unroll
    for (int j = 0; j < 4; ++j) {
        int n = nt * 64 + j * 16 + (lane & 15);
        if (n < MDIM) {
#pragma unroll
            for (int r = 0; r < 8; ++r) {
                int bb = bBase + wv * 16 + laneHi * 8 + r;
                C[((size_t)bb * G + g) * MDIM + n] = acc[j][r];
            }
        }
    }
}

// ---------------------------------------------------------------- stage 5 ---
__global__ __launch_bounds__(256)
void final_reduce(const float* __restrict__ T, const float* __restrict__ aht4,
                  const float* __restrict__ fm, float* __restrict__ out) {
    int b = blockIdx.x, tid = threadIdx.x;
    __shared__ float sn[256], sd[256];
    const float* t0 = T + (size_t)b * 2 * MDIM;
    float num = 0.f, den = 0.f;
    for (int m = tid; m < MDIM; m += 256) {
        float w = nonneg(aht4[m]);
        num += w * t0[m] * t0[MDIM + m];
        den += w;
    }
    sn[tid] = num; sd[tid] = den;
    __syncthreads();
    for (int s = 128; s > 0; s >>= 1) {
        if (tid < s) { sn[tid] += sn[tid + s]; sd[tid] += sd[tid + s]; }
        __syncthreads();
    }
    if (tid == 0) out[b] = __logf(sn[0] / sd[0]) - 32.f * fm[b];
}

// ---------------------------------------------------------------------------
extern "C" void kernel_launch(void* const* d_in, const int* in_sizes, int n_in,
                              void* d_out, int out_size, void* d_ws, size_t ws_size,
                              hipStream_t stream) {
    const float* X    = (const float*)d_in[0];
    const float* w0   = (const float*)d_in[1];
    const float* b0   = (const float*)d_in[2];
    const float* a0   = (const float*)d_in[3];
    const float* w1   = (const float*)d_in[4];
    const float* b1   = (const float*)d_in[5];
    const float* a1   = (const float*)d_in[6];
    const float* w2   = (const float*)d_in[7];
    const float* b2   = (const float*)d_in[8];
    const float* aht[5] = { (const float*)d_in[9],  (const float*)d_in[10],
                            (const float*)d_in[11], (const float*)d_in[12],
                            (const float*)d_in[13] };
    float* out = (float*)d_out;

    // workspace layout (float units)
    float* ws   = (float*)d_ws;
    float* P    = ws;                                   // 512*32*1000 f (dead
                                                        // after stage 2; its
                                                        // 65.5MB is reused for
                                                        // the 62.9MB B panels)
    float* bufA = P    + (size_t)BDIM * DDIM * MDIM;    // 512*16*1000 f
    float* bufB = bufA + (size_t)BDIM * 16 * MDIM;      // 512*16*1000 f
    float* inv  = bufB + (size_t)BDIM * 16 * MDIM;      // 30*1000 f
    float* fm   = inv  + 30 * MDIM;                     // 512 f
    unsigned short* Abf = (unsigned short*)(fm + 512);  // 512*16*1024 bf16
    unsigned short* panels = (unsigned short*)P;        // reuse of P region

    const int    gs[4]   = {16, 8, 4, 2};
    const int    ioff[4] = {0, 16 * MDIM, 24 * MDIM, 28 * MDIM};
    const size_t poff[4] = {0, 16777216, 25165824, 29360128}; // ushort units

    // stage 1: MLP phidots
    mlp_phidots<<<dim3((DDIM * MDIM + 255) / 256, BDIM), 256, 0, stream>>>(
        X, w0, b0, a0, w1, b1, a1, w2, b2, P);

    // stage 2: stabilizer + first product-pool (bufA = T0 [B,16,M] fp32)
    stab_t0<<<dim3(BDIM), 256, 0, stream>>>(P, bufA, fm);

    // stage 3: mixture column normalizers
    for (int l = 0; l < 4; ++l)
        colsum_inv<<<dim3((MDIM + 255) / 256, gs[l]), 256, 0, stream>>>(
            aht[l], inv + ioff[l]);

    // stage 3b: prepacked bf16 B panels (overwrites P — safe: stream order)
    for (int l = 0; l < 4; ++l)
        prep_b<<<dim3(16, 32, gs[l]), 256, 0, stream>>>(
            aht[l], inv + ioff[l], panels + poff[l]);

    // stage 4: HT levels; fp32 C ping-pongs bufA <-> bufB, bf16 A in Abf
    const float* srcs[4] = {bufA, bufB, bufA, bufB};
    float*       dsts[4] = {bufB, bufA, bufB, bufA};
    for (int l = 0; l < 4; ++l) {
        int G = gs[l];
        prep_a<<<dim3(BDIM * G * KPAD / 256), 256, 0, stream>>>(
            srcs[l], Abf, G, l ? 1 : 0);
        ht_gemm<<<dim3(16, BDIM / 128, G), 256, 0, stream>>>(
            Abf, panels + poff[l], dsts[l], G);
    }

    // stage 5: final pool + mixture + log (L3 output is in bufA)
    final_reduce<<<dim3(BDIM), 256, 0, stream>>>(bufA, aht[4], fm, out);
}
